// MLP_43130061586864
// MI455X (gfx1250) — matrix operand order repaired and checked by reading, not compiled, not run on hardware
//
#include <hip/hip_runtime.h>
#include <hip/hip_bf16.h>

typedef __attribute__((ext_vector_type(16))) _Float16 v16h;
typedef __attribute__((ext_vector_type(8)))  float    v8f;

__device__ __forceinline__ v8f wmma_f16(v16h a, v16h b, v8f c) {
  // D = A(16x32 f16) * B(32x16 f16) + C(16x16 f32)
  return __builtin_amdgcn_wmma_f32_16x16x32_f16(false, a, false, b, (short)0, c,
                                                false, false);
}

// Weight (A) fragment. M = (lane&15)+row_off = output feature, K interleaved:
// kA(j,h) = j + 16*h.  Augmented column K==in_dim holds the bias.
__device__ __forceinline__ v16h make_a(const float* __restrict__ W,
                                       const float* __restrict__ bias,
                                       int out_dim, int in_dim, int row_off,
                                       int lane) {
  int m = (lane & 15) + row_off;
  int h = lane >> 4;
  v16h a;
#pragma unroll
  for (int j = 0; j < 16; ++j) {
    int k = j + 16 * h;
    float v = 0.f;
    if (m < out_dim) {
      if (k < in_dim)       v = W[m * in_dim + k];
      else if (k == in_dim) v = bias[m];
    }
    a[j] = (_Float16)v;
  }
  return a;
}

// Build next-layer activation (B) fragment from previous D tile(s).
// B layout: lane holds column N=lane&15, element j <-> K = j + 16*h.
// Own D tiles at lane (h=lane>>4): dlo[g] = feature g+8h, dhi[g] = feature 16+g+8h.
// Features in the "other" 8-band live in the partner lane (lane ^ 16).
// ReLU applied; K==IN_DIM overridden to 1.0 (bias row); K>IN_DIM zeroed.
template <int IN_DIM>
__device__ __forceinline__ v16h activ_b(const v8f& dlo, const v8f& dhi, int h) {
  v16h b;
#pragma unroll
  for (int j = 0; j < 16; ++j) {
    float v0 = 0.f, v1 = 0.f;              // value for h==0 (K=j), h==1 (K=16+j)
    if (j <= IN_DIM) {                     // K0 = j needed?
      if (j < 8) v0 = dlo[j];              // own: feature j (h==0)
      else       v0 = __shfl_xor(dlo[j - 8], 16); // partner's feature 8+(j-8)
    }
    if (16 + j <= IN_DIM) {                // K1 = 16+j needed?
      if (j < 8) v1 = __shfl_xor(dhi[j], 16);     // partner's feature 16+j
      else       v1 = dhi[j - 8];          // own: feature 24+(j-8) (h==1)
    }
    float v = h ? v1 : v0;
    int   K = j + 16 * h;
    v = fmaxf(v, 0.f);
    if (K == IN_DIM) v = 1.0f;
    if (K >  IN_DIM) v = 0.0f;
    b[j] = (_Float16)v;
  }
  return b;
}

__global__ __launch_bounds__(256) void mlp_segsum_kernel(
    const float* __restrict__ t, const float* __restrict__ pos,
    const float* __restrict__ poi_t, const float* __restrict__ poi_pos,
    const int* __restrict__ batch,
    const float* __restrict__ W1, const float* __restrict__ b1,
    const float* __restrict__ W2, const float* __restrict__ b2,
    const float* __restrict__ W3, const float* __restrict__ b3,
    const float* __restrict__ W4, const float* __restrict__ b4,
    const float* __restrict__ W5, const float* __restrict__ b5,
    float* __restrict__ out, int Ntot) {
  const int lane = threadIdx.x & 31;
  const int h    = lane >> 4;

  // Per-wave weight fragments (loop invariant, tiny, L2-resident).
  v16h A1  = make_a(W1, b1, 10,  2,  0, lane);
  v16h A2l = make_a(W2, b2, 20, 10,  0, lane);
  v16h A2h = make_a(W2, b2, 20, 10, 16, lane);
  v16h A3  = make_a(W3, b3, 10, 20,  0, lane);
  v16h A4  = make_a(W4, b4,  5, 10,  0, lane);
  v16h A5  = make_a(W5, b5,  1,  5,  0, lane);

  const v8f zero8 = {};

  long long wave  = (long long)blockIdx.x * (blockDim.x >> 5) + (threadIdx.x >> 5);
  long long waves = (long long)gridDim.x * (blockDim.x >> 5);

  for (long long base = wave * 32; base < (long long)Ntot; base += waves * 32) {
    int   p  = (int)base + lane;           // one point per lane (N % 32 == 0)
    float tv = t[p];
    float px = pos[2 * p + 0];
    float py = pos[2 * p + 1];
    int   bi = batch[p];
    float dt = tv - poi_t[bi];
    float dx = px - poi_pos[2 * bi + 0];
    float dy = py - poi_pos[2 * bi + 1];
    float r2 = dx * dx + dy * dy;

    // Tile0 = points base+0..15 (lanes 0..15), Tile1 = base+16..31 (shuffled down).
    float dtu = __shfl(dt, (lane & 15) + 16);
    float r2u = __shfl(r2, (lane & 15) + 16);

    v16h B1a = {};
    v16h B1b = {};
    if (h == 0) {                          // K=0:diff_t, K=1:r2, K=2:bias one
      B1a[0] = (_Float16)dt;  B1a[1] = (_Float16)r2;  B1a[2] = (_Float16)1.0f;
      B1b[0] = (_Float16)dtu; B1b[1] = (_Float16)r2u; B1b[2] = (_Float16)1.0f;
    }

    // Layer 1: 2 -> 10
    v8f D1a = wmma_f16(A1, B1a, zero8);
    v8f D1b = wmma_f16(A1, B1b, zero8);
    // Layer 2: 10 -> 20 (two output tiles)
    v16h B2a = activ_b<10>(D1a, zero8, h);
    v16h B2b = activ_b<10>(D1b, zero8, h);
    v8f D2la = wmma_f16(A2l, B2a, zero8);
    v8f D2ha = wmma_f16(A2h, B2a, zero8);
    v8f D2lb = wmma_f16(A2l, B2b, zero8);
    v8f D2hb = wmma_f16(A2h, B2b, zero8);
    // Layer 3: 20 -> 10
    v16h B3a = activ_b<20>(D2la, D2ha, h);
    v16h B3b = activ_b<20>(D2lb, D2hb, h);
    v8f D3a = wmma_f16(A3, B3a, zero8);
    v8f D3b = wmma_f16(A3, B3b, zero8);
    // Layer 4: 10 -> 5
    v16h B4a = activ_b<10>(D3a, zero8, h);
    v16h B4b = activ_b<10>(D3b, zero8, h);
    v8f D4a = wmma_f16(A4, B4a, zero8);
    v8f D4b = wmma_f16(A4, B4b, zero8);
    // Layer 5: 5 -> 1
    v16h B5a = activ_b<5>(D4a, zero8, h);
    v16h B5b = activ_b<5>(D4b, zero8, h);
    v8f D5a = wmma_f16(A5, B5a, zero8);
    v8f D5b = wmma_f16(A5, B5b, zero8);

    // Weight for point N lives at lane N, VGPR0 (M=0). Tile1 -> upper lanes.
    float wb = __shfl(D5b[0], lane & 15);
    float w  = (lane < 16) ? D5a[0] : wb;

    float nrm = fmaxf(sqrtf(r2), 1e-12f);
    float s   = w / nrm;
    float cx  = s * dx;
    float cy  = s * dy;

    // batch is sorted: nearly every wave sits inside one segment.
    int bFirst = __shfl(bi, 0);
    int bLast  = __shfl(bi, 31);
    if (bFirst == bLast) {
#pragma unroll
      for (int off = 16; off > 0; off >>= 1) {
        cx += __shfl_xor(cx, off);
        cy += __shfl_xor(cy, off);
      }
      if (lane == 0) {
        atomicAdd(&out[2 * bFirst + 0], cx);
        atomicAdd(&out[2 * bFirst + 1], cy);
      }
    } else {
      atomicAdd(&out[2 * bi + 0], cx);
      atomicAdd(&out[2 * bi + 1], cy);
    }
  }
}

extern "C" void kernel_launch(void* const* d_in, const int* in_sizes, int n_in,
                              void* d_out, int out_size, void* d_ws, size_t ws_size,
                              hipStream_t stream) {
  (void)n_in; (void)d_ws; (void)ws_size;
  const float* t       = (const float*)d_in[0];
  const float* pos     = (const float*)d_in[1];
  const float* poi_t   = (const float*)d_in[2];
  const float* poi_pos = (const float*)d_in[3];
  const int*   batch   = (const int*)d_in[4];
  const float* W1 = (const float*)d_in[5];
  const float* b1 = (const float*)d_in[6];
  const float* W2 = (const float*)d_in[7];
  const float* b2 = (const float*)d_in[8];
  const float* W3 = (const float*)d_in[9];
  const float* b3 = (const float*)d_in[10];
  const float* W4 = (const float*)d_in[11];
  const float* b4 = (const float*)d_in[12];
  const float* W5 = (const float*)d_in[13];
  const float* b5 = (const float*)d_in[14];
  float* out = (float*)d_out;
  const int N = in_sizes[0];

  // Atomic accumulation target must start at zero every call (graph-safe memset).
  hipMemsetAsync(d_out, 0, (size_t)out_size * sizeof(float), stream);

  dim3 block(256);          // 8 waves/block (wave32)
  dim3 grid(2048);          // 16384 waves, 16 iterations of 32 points each
  mlp_segsum_kernel<<<grid, block, 0, stream>>>(
      t, pos, poi_t, poi_pos, batch,
      W1, b1, W2, b2, W3, b3, W4, b4, W5, b5, out, N);
}
